// AttnBlock_28810640622359
// MI455X (gfx1250) — matrix operand order repaired and checked by reading, not compiled, not run on hardware
//
#include <hip/hip_runtime.h>
#include <hip/hip_bf16.h>

typedef _Float16 f16;
typedef __attribute__((ext_vector_type(16))) _Float16 v16h;
typedef __attribute__((ext_vector_type(8)))  _Float16 v8h;
typedef __attribute__((ext_vector_type(8)))  float    v8f;

#define B_ 4
#define C_ 512
#define N_ 4096
#define G_ 32
#define CPG 16
#define SCALE_QK 0.04419417382415922f  /* 512^-0.5 */

// ---------------------------------------------------------------------------
// D = A(16x32 f16) x B(32x16 f16) + C(f32)
__device__ inline v8f wmma32(v16h a, v16h b, v8f c) {
  return __builtin_amdgcn_wmma_f32_16x16x32_f16(
      /*neg_a=*/false, a, /*neg_b=*/false, b,
      /*c_mod=*/(short)0, c, /*reuse_a=*/false, /*reuse_b=*/false);
}

// A-operand (M=lane%16 row, K contiguous at kk+half*16 .. +15) from a
// row-major [M][K-contig] f16 row pointer.
__device__ inline v16h load_a(const f16* __restrict__ row, int kk, int hlf) {
  v8h lo = *(const v8h*)(row + kk + hlf * 16);
  v8h hi = *(const v8h*)(row + kk + hlf * 16 + 8);
  v16h a;
#pragma unroll
  for (int e = 0; e < 8; ++e) { a[e] = lo[e]; a[e + 8] = hi[e]; }
  return a;
}

// ---------------------------------------------------------------------------
// Kernel 1: GroupNorm; writes hn transposed [b][n][c] as f16 (c contiguous).
__global__ void __launch_bounds__(256)
groupnorm_kernel(const float* __restrict__ x, const float* __restrict__ gamma,
                 const float* __restrict__ beta, f16* __restrict__ hn_t) {
  __shared__ float s1[256], s2[256];
  const int b = blockIdx.x / G_, g = blockIdx.x % G_;
  const int t = threadIdx.x;
  const float* xb = x + ((size_t)b * C_ + g * CPG) * N_;
  float sum = 0.f, sq = 0.f;
  for (int idx = t; idx < CPG * N_; idx += 256) {
    float v = xb[idx];
    sum += v; sq += v * v;
  }
  s1[t] = sum; s2[t] = sq; __syncthreads();
  for (int off = 128; off; off >>= 1) {
    if (t < off) { s1[t] += s1[t + off]; s2[t] += s2[t + off]; }
    __syncthreads();
  }
  const float mean = s1[0] * (1.0f / (CPG * N_));
  const float var  = s2[0] * (1.0f / (CPG * N_)) - mean * mean;
  const float rinv = rsqrtf(var + 1e-6f);
  for (int idx = t; idx < CPG * N_; idx += 256) {
    int cc = idx & (CPG - 1);
    int pix = idx >> 4;
    int ch = g * CPG + cc;
    float v = xb[(size_t)cc * N_ + pix];
    float y = (v - mean) * rinv * gamma[ch] + beta[ch];
    hn_t[((size_t)b * N_ + pix) * C_ + ch] = (f16)y;
  }
}

// ---------------------------------------------------------------------------
// Kernel 2: weight f32 -> f16 (row-major preserved).
__global__ void __launch_bounds__(256)
cvt_w_kernel(const float* __restrict__ w, f16* __restrict__ w16) {
  int i = blockIdx.x * 256 + threadIdx.x;
  w16[i] = (f16)w[i];
}

// ---------------------------------------------------------------------------
// Kernel 3: out[o][i] = W[o][c] * hn[c][i] + bias[o].
// One wave computes a 16(o) x 64(i) tile (4 accumulators, A reused).
// transposed=1: store as [b][n][o] f16 (o contiguous)  -> used for q, k
// transposed=0: store as [b][o][n] f16 (n contiguous)  -> used for v
__global__ void __launch_bounds__(256)
qkv_kernel(const f16* __restrict__ w16, const float* __restrict__ bias,
           const f16* __restrict__ hn_t, f16* __restrict__ out16,
           int transposed) {
  const int gw = blockIdx.x * 8 + (threadIdx.x >> 5);
  const int lane = threadIdx.x & 31;
  const int hlf = lane >> 4, l16 = lane & 15;
  const int it = gw & 63;           // 64 i-tiles of width 64
  const int ot = (gw >> 6) & 31;    // 32 o-tiles
  const int b  = gw >> 11;          // 4 batches
  const f16* arow = w16 + (size_t)(ot * 16 + l16) * C_;

  v8f acc[4] = {};
  for (int kk = 0; kk < C_; kk += 32) {
    v16h a = load_a(arow, kk, hlf);
#pragma unroll
    for (int n = 0; n < 4; ++n) {
      int i = it * 64 + n * 16 + l16;
      v16h bm = *(const v16h*)(hn_t + ((size_t)b * N_ + i) * C_ + kk + hlf * 16);
      acc[n] = wmma32(a, bm, acc[n]);
    }
  }
  float bv[8];
#pragma unroll
  for (int v = 0; v < 8; ++v) bv[v] = bias[ot * 16 + hlf * 8 + v];

  if (transposed) {
#pragma unroll
    for (int n = 0; n < 4; ++n) {
      int i = it * 64 + n * 16 + l16;
      v8h st;
#pragma unroll
      for (int v = 0; v < 8; ++v) st[v] = (f16)(acc[n][v] + bv[v]);
      *(v8h*)(out16 + ((size_t)b * N_ + i) * C_ + ot * 16 + hlf * 8) = st;
    }
  } else {
#pragma unroll
    for (int n = 0; n < 4; ++n) {
      int i = it * 64 + n * 16 + l16;
#pragma unroll
      for (int v = 0; v < 8; ++v)
        out16[((size_t)b * C_ + ot * 16 + hlf * 8 + v) * N_ + i] =
            (f16)(acc[n][v] + bv[v]);
    }
  }
}

// ---------------------------------------------------------------------------
// Kernel 4: attention for one (batch, 16-row query tile) per workgroup.
// Full 16x4096 score row kept in 256KB LDS (<= 320KB WGP LDS).
__global__ void __launch_bounds__(256)
attn_kernel(const f16* __restrict__ q_t,   // [b][n][c]
            const f16* __restrict__ k_t,   // [b][n][c]
            const f16* __restrict__ vm,    // [b][c][n]
            f16* __restrict__ att_t) {     // [b][n][c]
  __shared__ float S[16 * N_];             // 256 KB score/prob tile
  __shared__ float red[256];
  const int b  = blockIdx.x >> 8;          // N_/16 = 256 tiles per batch
  const int i0 = (blockIdx.x & 255) * 16;
  const int wv = threadIdx.x >> 5;
  const int lane = threadIdx.x & 31;
  const int hlf = lane >> 4, l16 = lane & 15;

  // Phase 1: S[i][j] = scale * sum_c q_t[i][c] * k_t[j][c]
  // Each wave computes 4 adjacent 16x16 j-tiles per pass: one A load per
  // k-step feeds 4 WMMAs (A-operand traffic amortized 4x).
  const f16* arow = q_t + ((size_t)b * N_ + i0 + l16) * C_;
  for (int jt0 = wv * 4; jt0 < N_ / 16; jt0 += 32) {
    const f16* brow0 = k_t + ((size_t)b * N_ + jt0 * 16 + l16) * C_;
    v8f acc[4] = {};
    for (int kk = 0; kk < C_; kk += 32) {
      v16h a = load_a(arow, kk, hlf);
#pragma unroll
      for (int n = 0; n < 4; ++n) {
        v16h bm = *(const v16h*)(brow0 + (size_t)n * 16 * C_ + kk + hlf * 16);
        acc[n] = wmma32(a, bm, acc[n]);
      }
    }
#pragma unroll
    for (int n = 0; n < 4; ++n)
#pragma unroll
      for (int v = 0; v < 8; ++v)
        S[(size_t)(v + hlf * 8) * N_ + (jt0 + n) * 16 + l16] =
            acc[n][v] * SCALE_QK;
  }
  __syncthreads();

  // Phase 2: row softmax (16 threads per row).
  {
    const int r = threadIdx.x >> 4, c = threadIdx.x & 15;
    float* row = S + (size_t)r * N_;
    float m = -1e30f;
    for (int j = c; j < N_; j += 16) m = fmaxf(m, row[j]);
    red[threadIdx.x] = m; __syncthreads();
    for (int off = 8; off; off >>= 1) {
      if (c < off) red[threadIdx.x] = fmaxf(red[threadIdx.x], red[threadIdx.x + off]);
      __syncthreads();
    }
    m = red[r * 16]; __syncthreads();
    float s = 0.f;
    for (int j = c; j < N_; j += 16) {
      float e = __expf(row[j] - m);
      row[j] = e; s += e;
    }
    red[threadIdx.x] = s; __syncthreads();
    for (int off = 8; off; off >>= 1) {
      if (c < off) red[threadIdx.x] += red[threadIdx.x + off];
      __syncthreads();
    }
    float inv = 1.0f / red[r * 16];
    for (int j = c; j < N_; j += 16) row[j] *= inv;
  }
  __syncthreads();

  // Phase 3: O[i][c] = sum_j P[i][j] * v[c][j]; wave handles 64 c-columns.
  const int c0 = wv * 64;
  v8f accO[4] = {};
  const float* prow = S + (size_t)l16 * N_;
  for (int jb = 0; jb < N_; jb += 32) {
    v16h a;
#pragma unroll
    for (int e = 0; e < 8; ++e) {
      a[e]     = (f16)prow[jb + hlf * 16 + e];
      a[e + 8] = (f16)prow[jb + hlf * 16 + 8 + e];
    }
#pragma unroll
    for (int n = 0; n < 4; ++n) {
      int c = c0 + n * 16 + l16;
      v16h bm = *(const v16h*)(vm + ((size_t)b * C_ + c) * N_ + jb + hlf * 16);
      accO[n] = wmma32(a, bm, accO[n]);
    }
  }
#pragma unroll
  for (int n = 0; n < 4; ++n)
#pragma unroll
    for (int v = 0; v < 8; ++v)
      att_t[((size_t)b * N_ + i0 + v + hlf * 8) * C_ + c0 + n * 16 + l16] =
          (f16)accO[n][v];
}

// ---------------------------------------------------------------------------
// Kernel 5: out = x + Wo * att + bo   (fp32 output).
__global__ void __launch_bounds__(256)
proj_kernel(const f16* __restrict__ wo16, const float* __restrict__ bo,
            const f16* __restrict__ att_t, const float* __restrict__ x,
            float* __restrict__ out) {
  const int gw = blockIdx.x * 8 + (threadIdx.x >> 5);
  const int lane = threadIdx.x & 31;
  const int hlf = lane >> 4, l16 = lane & 15;
  const int it = gw & 63, ot = (gw >> 6) & 31, b = gw >> 11;
  const f16* arow = wo16 + (size_t)(ot * 16 + l16) * C_;

  v8f acc[4] = {};
  for (int kk = 0; kk < C_; kk += 32) {
    v16h a = load_a(arow, kk, hlf);
#pragma unroll
    for (int n = 0; n < 4; ++n) {
      int i = it * 64 + n * 16 + l16;
      v16h bm = *(const v16h*)(att_t + ((size_t)b * N_ + i) * C_ + kk + hlf * 16);
      acc[n] = wmma32(a, bm, acc[n]);
    }
  }
  float bv[8];
#pragma unroll
  for (int v = 0; v < 8; ++v) bv[v] = bo[ot * 16 + hlf * 8 + v];
#pragma unroll
  for (int n = 0; n < 4; ++n) {
    int i = it * 64 + n * 16 + l16;
#pragma unroll
    for (int v = 0; v < 8; ++v) {
      size_t idx = ((size_t)b * C_ + ot * 16 + hlf * 8 + v) * N_ + i;
      out[idx] = x[idx] + acc[n][v] + bv[v];
    }
  }
}

// ---------------------------------------------------------------------------
extern "C" void kernel_launch(void* const* d_in, const int* in_sizes, int n_in,
                              void* d_out, int out_size, void* d_ws, size_t ws_size,
                              hipStream_t stream) {
  (void)in_sizes; (void)n_in; (void)out_size; (void)ws_size;
  const float* x     = (const float*)d_in[0];
  const float* gamma = (const float*)d_in[1];
  const float* beta  = (const float*)d_in[2];
  const float* wq = (const float*)d_in[3];
  const float* bq = (const float*)d_in[4];
  const float* wk = (const float*)d_in[5];
  const float* bk = (const float*)d_in[6];
  const float* wv = (const float*)d_in[7];
  const float* bv = (const float*)d_in[8];
  const float* wo = (const float*)d_in[9];
  const float* bo = (const float*)d_in[10];
  float* out = (float*)d_out;

  char* ws = (char*)d_ws;
  const size_t WSZ = (size_t)C_ * C_ * sizeof(f16);       // 512 KB
  const size_t MAT = (size_t)B_ * N_ * C_ * sizeof(f16);  // 16 MB
  f16* wq16 = (f16*)(ws + 0 * WSZ);
  f16* wk16 = (f16*)(ws + 1 * WSZ);
  f16* wv16 = (f16*)(ws + 2 * WSZ);
  f16* wo16 = (f16*)(ws + 3 * WSZ);
  size_t off = 4 * WSZ;
  f16* hn_t = (f16*)(ws + off); off += MAT;   // [b][n][c]
  f16* q_t  = (f16*)(ws + off); off += MAT;   // [b][n][c]
  f16* k_t  = (f16*)(ws + off); off += MAT;   // [b][n][c]
  f16* v_m  = (f16*)(ws + off); off += MAT;   // [b][c][n]
  f16* a_t  = (f16*)(ws + off); off += MAT;   // [b][n][c]

  groupnorm_kernel<<<B_ * G_, 256, 0, stream>>>(x, gamma, beta, hn_t);
  cvt_w_kernel<<<C_ * C_ / 256, 256, 0, stream>>>(wq, wq16);
  cvt_w_kernel<<<C_ * C_ / 256, 256, 0, stream>>>(wk, wk16);
  cvt_w_kernel<<<C_ * C_ / 256, 256, 0, stream>>>(wv, wv16);
  cvt_w_kernel<<<C_ * C_ / 256, 256, 0, stream>>>(wo, wo16);
  qkv_kernel<<<1024, 256, 0, stream>>>(wq16, bq, hn_t, q_t, 1);
  qkv_kernel<<<1024, 256, 0, stream>>>(wk16, bk, hn_t, k_t, 1);
  qkv_kernel<<<1024, 256, 0, stream>>>(wv16, bv, hn_t, v_m, 0);
  attn_kernel<<<B_ * (N_ / 16), 256, 0, stream>>>(q_t, k_t, v_m, a_t);
  proj_kernel<<<1024, 256, 0, stream>>>(wo16, bo, a_t, x, out);
}
